// GraphNeuralNetwork_49933289783572
// MI455X (gfx1250) — matrix operand order repaired and checked by reading, not compiled, not run on hardware
//
#include <hip/hip_runtime.h>
#include <hip/hip_bf16.h>

#define N_NODES_C  10000
#define N_EDGES_C  20000
#define N_GRAPHS_C 64
#define DIM_C      64
#define NFEAT_C    32
#define EDIM_C     16
#define NCLS_C     10
#define BN_EPS     1e-5f

typedef __attribute__((ext_vector_type(16))) _Float16 v16h;
typedef __attribute__((ext_vector_type(8)))  _Float16 v8h;
typedef __attribute__((ext_vector_type(8)))  float    v8f;

// Use the Tensor Data Mover for LDS staging when the toolchain exposes it.
#if defined(__gfx1250__) && __has_builtin(__builtin_amdgcn_tensor_load_to_lds) && \
    __has_builtin(__builtin_amdgcn_s_wait_tensorcnt)
#define USE_TDM 1
#else
#define USE_TDM 0
#endif

__device__ __forceinline__ v8f wmma_f32_f16(v16h a, v16h b, v8f c) {
  // 8 args: (neg_a, A, neg_b, B, c_mod, C, reuse_a, reuse_b)
  return __builtin_amdgcn_wmma_f32_16x16x32_f16(false, a, false, b, (short)0, c, false, false);
}

// ---- fragment loaders (layouts per CDNA5 ISA 7.12.2, wave32) ----

// A fragment: 16x32 f16 tile, row-major source with leading dim ld.
// lane holds row (lane&15); lanes 0-15 hold K {0..7,16..23}, lanes 16-31 hold K {8..15,24..31}.
__device__ __forceinline__ v16h load_a_frag(const _Float16* __restrict__ base, int ld, int lane) {
  int m  = lane & 15;
  int hi = (lane >> 4) & 1;
  const _Float16* p = base + (size_t)m * ld + hi * 8;
  v16h a;
#pragma unroll
  for (int j = 0; j < 8; ++j) a[j] = p[j];
#pragma unroll
  for (int j = 0; j < 8; ++j) a[8 + j] = p[16 + j];
  return a;
}

// B fragment: 32x16 f16 tile B[k][n], loaded from N-major (pre-transposed) weights wt[n][k].
// lane holds column (lane&15); lanes 0-15 hold K {0..7,16..23}, lanes 16-31 hold K {8..15,24..31}.
__device__ __forceinline__ v16h load_b_frag(const _Float16* __restrict__ wt, int ld, int row0, int kofs, int lane) {
  int n  = lane & 15;
  int hi = (lane >> 4) & 1;
  const _Float16* p = wt + (size_t)(row0 + n) * ld + kofs + hi * 8;
  v16h b;
#pragma unroll
  for (int j = 0; j < 8; ++j) { b[j] = p[j]; b[8 + j] = p[16 + j]; }
  return b;
}

#if USE_TDM
// ---- Tensor Data Mover: stage a 64-row x IN_DIM-col f16 tile (row pitch KEXT) into LDS ----
typedef __attribute__((ext_vector_type(4))) unsigned int tdm_u32x4;
typedef __attribute__((ext_vector_type(8))) int          tdm_i32x8;
typedef __attribute__((ext_vector_type(4))) int          tdm_i32x4;

template <int IN_DIM>
__device__ __forceinline__ void tdm_stage_slab(const _Float16* w2te, int c, _Float16* lds_dst) {
  constexpr unsigned KEXT = 128u * IN_DIM + IN_DIM;
  unsigned long long ga = (unsigned long long)(const void*)(w2te + (size_t)c * IN_DIM);
  // LDS flat addresses are aperture-high | byte-offset-low (ISA 10.2): low 32 bits = lds_addr.
  unsigned lds_off = (unsigned)(unsigned long long)(void*)lds_dst;

  tdm_u32x4 g0;
  g0.x = 1u;                                                    // count=1 (valid), user mode
  g0.y = lds_off;                                               // lds_addr (bytes)
  g0.z = (unsigned)(ga & 0xffffffffull);                        // global_addr[31:0]
  g0.w = (unsigned)((ga >> 32) & 0x01ffffffull) | (2u << 30);   // global_addr[56:32] | type=2

  tdm_i32x8 g1;
  g1[0] = (int)(1u << 16);                                      // wg_mask=0, data_size=1 (2B), no flags
  g1[1] = (int)(((unsigned)IN_DIM & 0xffffu) << 16);            // abar_addr=0 | tensor_dim0[15:0]
  g1[2] = (int)((((unsigned)IN_DIM >> 16) & 0xffffu) | (64u << 16)); // tensor_dim0[31:16] | tensor_dim1[15:0]=64
  g1[3] = (int)(((unsigned)IN_DIM & 0xffffu) << 16);            // tensor_dim1[31:16]=0 | tile_dim0=IN_DIM
  g1[4] = (int)(64u);                                           // tile_dim1=64 | tile_dim2=0
  g1[5] = (int)KEXT;                                            // tensor_dim0_stride[31:0] (elements)
  g1[6] = 0;                                                    // stride[47:32]=0 | dim1_stride[15:0]=0
  g1[7] = 0;                                                    // dim1_stride[47:16]=0

  tdm_i32x4 g2 = {0, 0, 0, 0};
  tdm_i32x4 g3 = {0, 0, 0, 0};
#if __clang_major__ >= 23
  tdm_i32x8 g4 = {0, 0, 0, 0, 0, 0, 0, 0};
  __builtin_amdgcn_tensor_load_to_lds(g0, g1, g2, g3, g4, 0);   // 6-arg form (clang-23 / therock)
#else
  __builtin_amdgcn_tensor_load_to_lds(g0, g1, g2, g3, 0);       // 5-arg form (ROCm 7.2 / clang-22)
#endif
}
#endif  // USE_TDM

// ---------------- small prep / elementwise kernels ----------------

__global__ void zero_kernel(float* __restrict__ p, int n) {
  int i = blockIdx.x * blockDim.x + threadIdx.x;
  if (i < n) p[i] = 0.f;
}

// e = edge_attr @ et_w + et_b   -> f16 [E, 64]
__global__ void edge_transform_kernel(const float* __restrict__ ea, const float* __restrict__ etw,
                                      const float* __restrict__ etb, _Float16* __restrict__ eF) {
  int idx = blockIdx.x * blockDim.x + threadIdx.x;
  if (idx >= N_EDGES_C * DIM_C) return;
  int e = idx / DIM_C, o = idx - e * DIM_C;
  float s = etb[o];
#pragma unroll
  for (int k = 0; k < EDIM_C; ++k) s += ea[(size_t)e * EDIM_C + k] * etw[k * DIM_C + o];
  eF[idx] = (_Float16)s;
}

// w1t[l][n][k] = w1_l[k][n]  (f16, N-major), 5 layers of [64,128] -> [128,64]
__global__ void prep_w1t_kernel(const float* __restrict__ nn1_w1, const float* __restrict__ convs_w1,
                                _Float16* __restrict__ w1t) {
  int idx = blockIdx.x * blockDim.x + threadIdx.x;
  if (idx >= 5 * 128 * 64) return;
  int l = idx / (128 * 64);
  int r = idx - l * (128 * 64);
  int n = r / 64, k = r - n * 64;
  float v = (l == 0) ? nn1_w1[k * 128 + n]
                     : convs_w1[((size_t)(l - 1) * 64 + k) * 128 + n];
  w1t[idx] = (_Float16)v;
}

// w2te[n][k] for k < 128*in: w2[c][i*64+n] with c=k/in, i=k%in  (N-major flattened GEMM-B)
// for k >= 128*in (bias region): b2[i*64+n], i = k - 128*in
__global__ void prep_w2te_kernel(const float* __restrict__ w2, const float* __restrict__ b2,
                                 _Float16* __restrict__ w2te, int in_dim) {
  int kext  = 128 * in_dim + in_dim;
  int total = 64 * kext;
  int idx = blockIdx.x * blockDim.x + threadIdx.x;
  if (idx >= total) return;
  int n = idx / kext, k = idx - n * kext;
  float v;
  if (k < 128 * in_dim) {
    int c = k / in_dim, i = k - c * in_dim;
    v = w2[(size_t)c * (in_dim * 64) + i * 64 + n];
  } else {
    int i = k - 128 * in_dim;
    v = b2[i * 64 + n];
  }
  w2te[idx] = (_Float16)v;
}

// xs[e][i] = (f16) hsrc[src[e]][i]
__global__ void gather_kernel(const float* __restrict__ hsrc, const int* __restrict__ src,
                              _Float16* __restrict__ xs, int in_dim) {
  int idx = blockIdx.x * blockDim.x + threadIdx.x;
  if (idx >= N_EDGES_C * in_dim) return;
  int e = idx / in_dim, i = idx - e * in_dim;
  xs[idx] = (_Float16)hsrc[(size_t)src[e] * in_dim + i];
}

// ---------------- WMMA GEMM1: h1 = relu(e @ w1 + b1), [E,64]x[64,128] ----------------

__global__ void __launch_bounds__(256)
gemm1_kernel(const _Float16* __restrict__ eF, const _Float16* __restrict__ w1t,
             const float* __restrict__ b1, _Float16* __restrict__ h1) {
  const int lane = threadIdx.x & 31;
  const int wave = (blockIdx.x * 256 + threadIdx.x) >> 5;  // one 16-edge M-tile per wave
  if (wave * 16 >= N_EDGES_C) return;                      // wave-uniform exit (EXEC all-1 for WMMA)

  const _Float16* Ab = eF + (size_t)wave * 16 * DIM_C;
  v16h a0 = load_a_frag(Ab + 0,  DIM_C, lane);
  v16h a1 = load_a_frag(Ab + 32, DIM_C, lane);
  const int n = lane & 15, hi = lane >> 4;

#pragma unroll
  for (int nt = 0; nt < 8; ++nt) {
    v8f c;
#pragma unroll
    for (int r = 0; r < 8; ++r) c[r] = 0.f;
    v16h b0  = load_b_frag(w1t, DIM_C, nt * 16, 0,  lane);
    v16h b1f = load_b_frag(w1t, DIM_C, nt * 16, 32, lane);
    c = wmma_f32_f16(a0, b0, c);
    c = wmma_f32_f16(a1, b1f, c);
    float bias = b1[nt * 16 + n];
#pragma unroll
    for (int r = 0; r < 8; ++r) {
      float v = c[r] + bias;
      v = v > 0.f ? v : 0.f;
      h1[(size_t)(wave * 16 + r + 8 * hi) * 128 + nt * 16 + n] = (_Float16)v;
    }
  }
}

// ---------------- fused outer-product GEMM2 + bmm + bias + scatter-add ----------------
// msg[E,64] = (h1 outer x_src)[E, 128*IN] @ w2v[128*IN,64]  (+ x_src @ b2mat via extra c-step)
// A K-chunks are built on the fly: scalar h1[e,c] * cached x_src fragment.
// B slab (64 x IN_DIM f16, current c) is staged into LDS: TDM double-buffered DMA when
// available (tensor_load_to_lds + s_wait_tensorcnt), else synchronous vector copies.

template <int IN_DIM>
__global__ void __launch_bounds__(256)
msg_kernel(const _Float16* __restrict__ h1, const _Float16* __restrict__ w2te,
           const _Float16* __restrict__ xsrc, const int* __restrict__ dst,
           float* __restrict__ agg) {
  constexpr int KEXT = 128 * IN_DIM + IN_DIM;
  constexpr int NCH  = IN_DIM / 32;     // K-chunks of 32 per c
  constexpr int SLAB = 64 * IN_DIM;     // f16 elements per buffer
  __shared__ __align__(16) _Float16 Bslab[2 * SLAB];

  const int lane = threadIdx.x & 31;
  const int wv   = threadIdx.x >> 5;
  const int tile = blockIdx.x * 8 + wv;                 // 16-edge M-tile per wave
  const bool active = (tile * 16) < N_EDGES_C;          // wave-uniform

  v16h xf[NCH];
  if (active) {
    const _Float16* xb = xsrc + (size_t)tile * 16 * IN_DIM;
#pragma unroll
    for (int ch = 0; ch < NCH; ++ch) xf[ch] = load_a_frag(xb + ch * 32, IN_DIM, lane);
  }

  v8f acc[4];
#pragma unroll
  for (int nt = 0; nt < 4; ++nt)
#pragma unroll
    for (int r = 0; r < 8; ++r) acc[nt][r] = 0.f;

  const _Float16* h1row = h1 + (size_t)(tile * 16 + (lane & 15)) * 128;

#if USE_TDM
  if (wv == 0) tdm_stage_slab<IN_DIM>(w2te, 0, Bslab);  // prologue: DMA stage c=0 into buf 0
#endif

  for (int c = 0; c <= 128; ++c) {   // c==128 is the folded-bias step (scalar = 1, B = b2^T)
    _Float16* slab = Bslab + (c & 1) * SLAB;
#if USE_TDM
    if (c < 128) {
      if (wv == 0) tdm_stage_slab<IN_DIM>(w2te, c + 1, Bslab + ((c + 1) & 1) * SLAB);
      __builtin_amdgcn_s_wait_tensorcnt(1);   // stage c complete (newest DMA may remain in flight)
    } else {
      __builtin_amdgcn_s_wait_tensorcnt(0);
    }
    __syncthreads();
#else
    __syncthreads();
    for (int idx = threadIdx.x * 8; idx < SLAB; idx += 256 * 8) {
      int nrow = idx / IN_DIM;
      int col  = idx - nrow * IN_DIM;
      *(v8h*)&slab[idx] = *(const v8h*)&w2te[(size_t)nrow * KEXT + (size_t)c * IN_DIM + col];
    }
    __syncthreads();
#endif

    if (active) {
      _Float16 hs = (c < 128) ? h1row[c] : (_Float16)1.0f;
#pragma unroll
      for (int ch = 0; ch < NCH; ++ch) {
        v16h a;
#pragma unroll
        for (int j = 0; j < 16; ++j) a[j] = xf[ch][j] * hs;   // on-the-fly outer-product A frag
#pragma unroll
        for (int nt = 0; nt < 4; ++nt) {
          v16h b = load_b_frag(slab, IN_DIM, nt * 16, ch * 32, lane);
          acc[nt] = wmma_f32_f16(a, b, acc[nt]);
        }
      }
    }

#if USE_TDM
    __syncthreads();   // next iteration's DMA reuses this buffer: all waves must be done reading
#endif
  }

  if (active) {
    const int hi = lane >> 4, n = lane & 15;
#pragma unroll
    for (int nt = 0; nt < 4; ++nt)
#pragma unroll
      for (int r = 0; r < 8; ++r) {
        int m = r + 8 * hi;
        int e = tile * 16 + m;
        atomicAdd(&agg[(size_t)dst[e] * DIM_C + nt * 16 + n], acc[nt][r]);
      }
  }
}

// ---------------- node update, BN, pool, head ----------------

__global__ void node_update_kernel(const float* __restrict__ agg, const float* __restrict__ hin,
                                   const float* __restrict__ root, const float* __restrict__ bias,
                                   float* __restrict__ y, int in_dim) {
  int idx = blockIdx.x * blockDim.x + threadIdx.x;
  if (idx >= N_NODES_C * DIM_C) return;
  int nd = idx / DIM_C, o = idx - nd * DIM_C;
  float s = agg[idx] + bias[o];
  for (int i = 0; i < in_dim; ++i) s += hin[(size_t)nd * in_dim + i] * root[i * DIM_C + o];
  y[idx] = s;
}

__global__ void bn_stats_kernel(const float* __restrict__ y, float* __restrict__ stats) {
  int o = blockIdx.x;  // 64 blocks, one per column
  float s = 0.f, s2 = 0.f;
  for (int nd = threadIdx.x; nd < N_NODES_C; nd += blockDim.x) {
    float v = y[(size_t)nd * DIM_C + o];
    s += v; s2 += v * v;
  }
  __shared__ float sh[512];
  sh[threadIdx.x] = s; sh[256 + threadIdx.x] = s2;
  __syncthreads();
  for (int st = 128; st > 0; st >>= 1) {
    if ((int)threadIdx.x < st) {
      sh[threadIdx.x] += sh[threadIdx.x + st];
      sh[256 + threadIdx.x] += sh[256 + threadIdx.x + st];
    }
    __syncthreads();
  }
  if (threadIdx.x == 0) {
    float m = sh[0] / (float)N_NODES_C;
    float v = sh[256] / (float)N_NODES_C - m * m;  // biased variance (matches reference)
    stats[o] = m; stats[64 + o] = v;
  }
}

__global__ void bn_apply_kernel(const float* __restrict__ y, const float* __restrict__ stats,
                                const float* __restrict__ gamma, const float* __restrict__ beta,
                                const float* __restrict__ hres, float* __restrict__ hout) {
  int idx = blockIdx.x * blockDim.x + threadIdx.x;
  if (idx >= N_NODES_C * DIM_C) return;
  int o = idx % DIM_C;
  float m = stats[o], v = stats[64 + o];
  float t = gamma[o] * (y[idx] - m) * rsqrtf(v + BN_EPS) + beta[o];
  t = t > 0.f ? t : 0.f;
  if (hres) t += hres[idx];
  hout[idx] = t;
}

// segment_max via int-bit atomicMax (all features are >= 0 post-ReLU; g pre-zeroed)
__global__ void pool_kernel(const float* __restrict__ h, const int* __restrict__ batch,
                            float* __restrict__ g) {
  int idx = blockIdx.x * blockDim.x + threadIdx.x;
  if (idx >= N_NODES_C * DIM_C) return;
  int nd = idx / DIM_C, o = idx - nd * DIM_C;
  atomicMax((int*)&g[(size_t)batch[nd] * DIM_C + o], __float_as_int(h[idx]));
}

__global__ void head_kernel(const float* __restrict__ g,
                            const float* __restrict__ w1, const float* __restrict__ b1,
                            const float* __restrict__ w2, const float* __restrict__ b2,
                            const float* __restrict__ wo, const float* __restrict__ bo,
                            float* __restrict__ out) {
  int gr = blockIdx.x;   // one block per graph, 64 threads
  int t  = threadIdx.x;
  __shared__ float cur[64], nxt[64];
  cur[t] = g[(size_t)gr * DIM_C + t];
  __syncthreads();
  float s = b1[t];
  for (int i = 0; i < 64; ++i) s += cur[i] * w1[i * 64 + t];
  nxt[t] = s > 0.f ? s : 0.f;
  __syncthreads();
  cur[t] = nxt[t];
  __syncthreads();
  s = b2[t];
  for (int i = 0; i < 64; ++i) s += cur[i] * w2[i * 64 + t];
  nxt[t] = s > 0.f ? s : 0.f;
  __syncthreads();
  cur[t] = nxt[t];
  __syncthreads();
  if (t < NCLS_C) {
    s = bo[t];
    for (int i = 0; i < 64; ++i) s += cur[i] * wo[i * NCLS_C + t];
    out[(size_t)gr * NCLS_C + t] = 1.f / (1.f + expf(-s));
  }
}

// ---------------- host orchestration ----------------

extern "C" void kernel_launch(void* const* d_in, const int* in_sizes, int n_in,
                              void* d_out, int out_size, void* d_ws, size_t ws_size,
                              hipStream_t stream) {
  const float* x          = (const float*)d_in[0];
  const int*   eidx       = (const int*)d_in[1];
  const float* edge_attr  = (const float*)d_in[2];
  const int*   batch      = (const int*)d_in[3];
  const float* et_w       = (const float*)d_in[4];
  const float* et_b       = (const float*)d_in[5];
  const float* nn1_w1     = (const float*)d_in[6];
  const float* nn1_b1     = (const float*)d_in[7];
  const float* nn1_w2     = (const float*)d_in[8];
  const float* nn1_b2     = (const float*)d_in[9];
  const float* root_in    = (const float*)d_in[10];
  const float* bias_in    = (const float*)d_in[11];
  const float* convs_w1   = (const float*)d_in[12];
  const float* convs_b1   = (const float*)d_in[13];
  const float* convs_w2   = (const float*)d_in[14];
  const float* convs_b2   = (const float*)d_in[15];
  const float* convs_root = (const float*)d_in[16];
  const float* convs_bias = (const float*)d_in[17];
  const float* bn_gamma   = (const float*)d_in[18];
  const float* bn_beta    = (const float*)d_in[19];
  const float* lin1_w     = (const float*)d_in[20];
  const float* lin1_b     = (const float*)d_in[21];
  const float* lin2_w     = (const float*)d_in[22];
  const float* lin2_b     = (const float*)d_in[23];
  const float* lout_w     = (const float*)d_in[24];
  const float* lout_b     = (const float*)d_in[25];

  char* p = (char*)d_ws;
  auto carve = [&](size_t bytes) -> void* {
    void* r = (void*)p;
    p += (bytes + 255) & ~(size_t)255;
    return r;
  };

  const size_t KEXT32 = 128 * 32 + 32;   // 4128
  const size_t KEXT64 = 128 * 64 + 64;   // 8256

  _Float16* eF    = (_Float16*)carve((size_t)N_EDGES_C * DIM_C * 2);
  _Float16* w1t   = (_Float16*)carve((size_t)5 * 128 * 64 * 2);
  _Float16* w2te0 = (_Float16*)carve((size_t)64 * KEXT32 * 2);
  _Float16* w2teR = (_Float16*)carve((size_t)4 * 64 * KEXT64 * 2);
  _Float16* xs    = (_Float16*)carve((size_t)N_EDGES_C * 64 * 2);
  _Float16* h1    = (_Float16*)carve((size_t)N_EDGES_C * 128 * 2);
  float* agg   = (float*)carve((size_t)N_NODES_C * DIM_C * 4);
  float* ybuf  = (float*)carve((size_t)N_NODES_C * DIM_C * 4);
  float* hA    = (float*)carve((size_t)N_NODES_C * DIM_C * 4);
  float* hB    = (float*)carve((size_t)N_NODES_C * DIM_C * 4);
  float* stats = (float*)carve(512);
  float* g     = (float*)carve((size_t)N_GRAPHS_C * DIM_C * 4);

  const int* src = eidx;
  const int* dst = eidx + N_EDGES_C;

  // one-time preps
  edge_transform_kernel<<<(N_EDGES_C * DIM_C + 255) / 256, 256, 0, stream>>>(edge_attr, et_w, et_b, eF);
  prep_w1t_kernel<<<(5 * 128 * 64 + 255) / 256, 256, 0, stream>>>(nn1_w1, convs_w1, w1t);
  prep_w2te_kernel<<<((int)(64 * KEXT32) + 255) / 256, 256, 0, stream>>>(nn1_w2, nn1_b2, w2te0, 32);
  for (int l = 0; l < 4; ++l)
    prep_w2te_kernel<<<((int)(64 * KEXT64) + 255) / 256, 256, 0, stream>>>(
        convs_w2 + (size_t)l * 128 * 4096, convs_b2 + (size_t)l * 4096,
        w2teR + (size_t)l * 64 * KEXT64, 64);

  const int nTiles    = N_EDGES_C / 16;          // 1250 (exact)
  const int msgBlocks = (nTiles + 7) / 8;        // 157
  const int g1Blocks  = (nTiles * 32 + 255) / 256;

  float* hcur = hA;
  float* hnext = hB;

  for (int l = 0; l < 5; ++l) {
    const int in_dim = (l == 0) ? NFEAT_C : DIM_C;
    const float* hin = (l == 0) ? x : hcur;
    gather_kernel<<<(N_EDGES_C * in_dim + 255) / 256, 256, 0, stream>>>(hin, src, xs, in_dim);

    const float* b1p = (l == 0) ? nn1_b1 : convs_b1 + (size_t)(l - 1) * 128;
    gemm1_kernel<<<g1Blocks, 256, 0, stream>>>(eF, w1t + (size_t)l * 128 * 64, b1p, h1);

    zero_kernel<<<(N_NODES_C * DIM_C + 255) / 256, 256, 0, stream>>>(agg, N_NODES_C * DIM_C);
    if (l == 0)
      msg_kernel<32><<<msgBlocks, 256, 0, stream>>>(h1, w2te0, xs, dst, agg);
    else
      msg_kernel<64><<<msgBlocks, 256, 0, stream>>>(h1, w2teR + (size_t)(l - 1) * 64 * KEXT64, xs, dst, agg);

    const float* rootp = (l == 0) ? root_in : convs_root + (size_t)(l - 1) * 64 * 64;
    const float* biasp = (l == 0) ? bias_in : convs_bias + (size_t)(l - 1) * 64;
    node_update_kernel<<<(N_NODES_C * DIM_C + 255) / 256, 256, 0, stream>>>(agg, hin, rootp, biasp, ybuf, in_dim);

    bn_stats_kernel<<<64, 256, 0, stream>>>(ybuf, stats);
    bn_apply_kernel<<<(N_NODES_C * DIM_C + 255) / 256, 256, 0, stream>>>(
        ybuf, stats, bn_gamma + (size_t)l * 64, bn_beta + (size_t)l * 64,
        (l == 0) ? (const float*)nullptr : hcur, hnext);

    float* t = hcur; hcur = hnext; hnext = t;
  }

  zero_kernel<<<(N_GRAPHS_C * DIM_C + 255) / 256, 256, 0, stream>>>(g, N_GRAPHS_C * DIM_C);
  pool_kernel<<<(N_NODES_C * DIM_C + 255) / 256, 256, 0, stream>>>(hcur, batch, g);
  head_kernel<<<N_GRAPHS_C, 64, 0, stream>>>(g, lin1_w, lin1_b, lin2_w, lin2_b, lout_w, lout_b, (float*)d_out);
}
